// Decoder_12893491822628
// MI455X (gfx1250) — compile-verified, matrix operand order checked by reading, not run on hardware
//
#include <hip/hip_runtime.h>
#include <hip/hip_bf16.h>
#include <math.h>

// Problem dims (fixed by reference)
#define B_  64
#define S_  64
#define H_  512
#define E_  512
#define T_  20
#define V_  32000
#define L_  3
#define NH_ 8
#define HD_ 64

typedef __attribute__((ext_vector_type(16))) __bf16 v16bf;
typedef __attribute__((ext_vector_type(8)))  __bf16 bf16x8;
typedef __attribute__((ext_vector_type(8)))  float  v8f;

// Load one 16-element bf16 fragment for the documented 16-bit A/B WMMA layout.
// p points at row start + (lane>>4)*8; elements 0..7 = K[c..c+7],
// 8..15 = K[c+16..c+23] (row length >= 32 elements, k-contiguous).
__device__ __forceinline__ v16bf load_frag(const __bf16* p) {
  bf16x8 lo = *(const bf16x8*)(p);
  bf16x8 hi = *(const bf16x8*)(p + 16);
  return __builtin_shufflevector(lo, hi, 0,1,2,3,4,5,6,7,8,9,10,11,12,13,14,15);
}

// ---------------------------------------------------------------------------
// Direct-load GEMM: one wave computes a 16(M) x 64(N) tile.
// Used for the small latency-bound per-timestep GEMMs.
// C[M,N] = A[M,K] @ Bw[N,K]^T + bias (bf16 in, fp32 accum), optional bf16 mirror.
// ---------------------------------------------------------------------------
__global__ void gemm_bf16_wmma(const __bf16* __restrict__ A, int lda,
                               const __bf16* __restrict__ Bw, int ldb,
                               const float* __restrict__ bias,
                               float* __restrict__ C, int ldc,
                               __bf16* __restrict__ Cb, int ldcb,
                               int K) {
  const int lane = threadIdx.x;
  const int half = lane >> 4;
  const int r    = lane & 15;
  const int m0 = blockIdx.y << 4;
  const int n0 = blockIdx.x << 6;

  v8f acc0 = {}, acc1 = {}, acc2 = {}, acc3 = {};

  const __bf16* ap = A  + (size_t)(m0 + r) * lda + half * 8;
  const __bf16* bp = Bw + (size_t)(n0 + r) * ldb + half * 8;
  const size_t bs = (size_t)ldb * 16;

  for (int k = 0; k < K; k += 32) {
    v16bf a  = load_frag(ap + k);
    v16bf b0 = load_frag(bp + k);
    v16bf b1 = load_frag(bp + bs + k);
    v16bf b2 = load_frag(bp + 2 * bs + k);
    v16bf b3 = load_frag(bp + 3 * bs + k);
    __builtin_prefetch(bp + k + 128, 0, 1);
    acc0 = __builtin_amdgcn_wmma_f32_16x16x32_bf16(false, a, false, b0, (short)0, acc0, false, false);
    acc1 = __builtin_amdgcn_wmma_f32_16x16x32_bf16(false, a, false, b1, (short)0, acc1, false, false);
    acc2 = __builtin_amdgcn_wmma_f32_16x16x32_bf16(false, a, false, b2, (short)0, acc2, false, false);
    acc3 = __builtin_amdgcn_wmma_f32_16x16x32_bf16(false, a, false, b3, (short)0, acc3, false, false);
  }

  const int c0 = n0 + r;
#pragma unroll
  for (int g = 0; g < 8; ++g) {
    const int row = m0 + g + half * 8;
    float v0 = acc0[g] + (bias ? bias[c0]      : 0.f);
    float v1 = acc1[g] + (bias ? bias[c0 + 16] : 0.f);
    float v2 = acc2[g] + (bias ? bias[c0 + 32] : 0.f);
    float v3 = acc3[g] + (bias ? bias[c0 + 48] : 0.f);
    float* cr = C + (size_t)row * ldc + c0;
    cr[0] = v0; cr[16] = v1; cr[32] = v2; cr[48] = v3;
    if (Cb) {
      __bf16* cb = Cb + (size_t)row * ldcb + c0;
      cb[0] = (__bf16)v0; cb[16] = (__bf16)v1;
      cb[32] = (__bf16)v2; cb[48] = (__bf16)v3;
    }
  }
}

// ---------------------------------------------------------------------------
// Async-staged GEMM: 4 waves (128 threads) compute a 64(M) x 64(N) tile.
// The shared 64x32 B tile is double-buffered in LDS via
// GLOBAL_LOAD_ASYNC_TO_LDS_B128 (ASYNCcnt), cutting B-operand request traffic
// 4x vs per-wave direct loads. Used for the dominant GEMMs (fc_w, K/V proj).
// Requires M % 64 == 0, N % 64 == 0, K % 32 == 0.
// ---------------------------------------------------------------------------
__global__ void gemm_bf16_wmma_async(const __bf16* __restrict__ A, int lda,
                                     const __bf16* __restrict__ Bw, int ldb,
                                     const float* __restrict__ bias,
                                     float* __restrict__ C, int ldc,
                                     int K) {
  __shared__ __bf16 smem[2][64 * 32];          // two 4KB B-tile buffers

  const int tid  = threadIdx.x;                 // 0..127
  const int wave = tid >> 5;                    // M sub-tile (0..3)
  const int lane = tid & 31;
  const int half = lane >> 4;
  const int r    = lane & 15;
  const int m0 = (blockIdx.y << 6) + (wave << 4);
  const int n0 = blockIdx.x << 6;

  // Staging assignment: thread -> (row = tid>>1, 16-element chunk = tid&1)
  const int srow = tid >> 1;
  const int schk = tid & 1;
  const __bf16* sgp = Bw + (size_t)(n0 + srow) * ldb + schk * 16;
  const unsigned sl0 = (unsigned)(size_t)(&smem[0][srow * 32 + schk * 16]);
  const unsigned sl1 = (unsigned)(size_t)(&smem[1][srow * 32 + schk * 16]);

  auto stage = [&](int k0, int buf) {
    unsigned long long ga = (unsigned long long)(size_t)(sgp + k0);
    unsigned l = buf ? sl1 : sl0;
    // two b128 async copies per thread: 128 thr * 32B = 4KB tile
    asm volatile("global_load_async_to_lds_b128 %0, %1, off"
                 :: "v"(l), "v"(ga) : "memory");
    asm volatile("global_load_async_to_lds_b128 %0, %1, off offset:16"
                 :: "v"(l + 16u), "v"(ga) : "memory");
  };

  auto ldfrag = [&](int buf, int n) -> v16bf {
    const __bf16* p = &smem[buf][n * 32 + half * 8];
    bf16x8 lo = *(const bf16x8*)p;
    bf16x8 hi = *(const bf16x8*)(p + 16);
    return __builtin_shufflevector(lo, hi, 0,1,2,3,4,5,6,7,8,9,10,11,12,13,14,15);
  };

  v8f acc0 = {}, acc1 = {}, acc2 = {}, acc3 = {};
  const __bf16* ap = A + (size_t)(m0 + r) * lda + half * 8;

  stage(0, 0);
  asm volatile("s_wait_asynccnt 0x0" ::: "memory");
  __syncthreads();

  const int nk = K >> 5;
  for (int i = 0; i < nk; ++i) {
    const int k0  = i << 5;
    const int cur = i & 1;
    if (i + 1 < nk) stage(k0 + 32, cur ^ 1);    // overlap next tile copy

    v16bf a  = load_frag(ap + k0);              // A rows: tiny, L0-resident
    v16bf b0 = ldfrag(cur, r);
    v16bf b1 = ldfrag(cur, r + 16);
    v16bf b2 = ldfrag(cur, r + 32);
    v16bf b3 = ldfrag(cur, r + 48);
    acc0 = __builtin_amdgcn_wmma_f32_16x16x32_bf16(false, a, false, b0, (short)0, acc0, false, false);
    acc1 = __builtin_amdgcn_wmma_f32_16x16x32_bf16(false, a, false, b1, (short)0, acc1, false, false);
    acc2 = __builtin_amdgcn_wmma_f32_16x16x32_bf16(false, a, false, b2, (short)0, acc2, false, false);
    acc3 = __builtin_amdgcn_wmma_f32_16x16x32_bf16(false, a, false, b3, (short)0, acc3, false, false);

    // next-buffer copies done (per-wave), then all waves sync before reuse
    asm volatile("s_wait_asynccnt 0x0" ::: "memory");
    __syncthreads();
  }

  const int c0 = n0 + r;
#pragma unroll
  for (int g = 0; g < 8; ++g) {
    const int row = m0 + g + half * 8;
    float v0 = acc0[g] + (bias ? bias[c0]      : 0.f);
    float v1 = acc1[g] + (bias ? bias[c0 + 16] : 0.f);
    float v2 = acc2[g] + (bias ? bias[c0 + 32] : 0.f);
    float v3 = acc3[g] + (bias ? bias[c0 + 48] : 0.f);
    float* cr = C + (size_t)row * ldc + c0;
    cr[0] = v0; cr[16] = v1; cr[32] = v2; cr[48] = v3;
  }
}

__global__ void cvt_f32_bf16(const float* __restrict__ src,
                             __bf16* __restrict__ dst, int n) {
  int i = blockIdx.x * blockDim.x + threadIdx.x;
  if (i < n) dst[i] = (__bf16)src[i];
}

__global__ void zero_state(float* __restrict__ hf, __bf16* __restrict__ hb, int n) {
  int i = blockIdx.x * blockDim.x + threadIdx.x;
  if (i < n) { hf[i] = 0.f; hb[i] = (__bf16)0.f; }
}

// x_bf[b,e] = bf16(embed_table[captions[b,t], e])
__global__ void embed_gather(const float* __restrict__ tbl,
                             const int* __restrict__ caps,
                             __bf16* __restrict__ xb, int t) {
  int i = blockIdx.x * blockDim.x + threadIdx.x;     // over B*E
  int b = i >> 9;
  int e = i & (E_ - 1);
  int tok = caps[b * T_ + t];
  xb[i] = (__bf16)tbl[(size_t)tok * E_ + e];
}

// PyTorch GRUCell gate math; gi/gh already include biases (GEMM epilogue).
__global__ void gru_gate(const float* __restrict__ gi,
                         const float* __restrict__ gh,
                         float* __restrict__ hf,
                         __bf16* __restrict__ hb) {
  int i = blockIdx.x * blockDim.x + threadIdx.x;     // over B*H
  int b = i >> 9;
  int c = i & (H_ - 1);
  size_t base = (size_t)b * (3 * H_);
  float ir = gi[base + c], iz = gi[base + H_ + c], inn = gi[base + 2 * H_ + c];
  float hr = gh[base + c], hz = gh[base + H_ + c], hnn = gh[base + 2 * H_ + c];
  float rr = 1.f / (1.f + expf(-(ir + hr)));
  float zz = 1.f / (1.f + expf(-(iz + hz)));
  float nn = tanhf(inn + rr * hnn);
  float hnew = (1.f - zz) * nn + zz * hf[i];
  hf[i] = hnew;
  hb[i] = (__bf16)hnew;
}

// One block (64 threads) per (batch, head): scores -> softmax -> context.
__global__ void attn_softmax_ctx(const float* __restrict__ q,
                                 const float* __restrict__ Kh,
                                 const float* __restrict__ Vh,
                                 __bf16* __restrict__ ctxb) {
  __shared__ float sc[S_];
  __shared__ float red[2];
  const int b  = blockIdx.x / NH_;
  const int nh = blockIdx.x % NH_;
  const int tid = threadIdx.x;                        // 0..63 (= S_ = HD_)

  const float* qp = q  + (size_t)b * H_ + nh * HD_;
  const float* kp = Kh + ((size_t)b * S_ + tid) * H_ + nh * HD_;
  float dot = 0.f;
#pragma unroll 8
  for (int d = 0; d < HD_; ++d) dot += qp[d] * kp[d];
  sc[tid] = dot * 0.125f;                             // 1/sqrt(64)
  __syncthreads();
  if (tid == 0) {
    float mx = sc[0];
    for (int s = 1; s < S_; ++s) mx = fmaxf(mx, sc[s]);
    red[0] = mx;
  }
  __syncthreads();
  sc[tid] = expf(sc[tid] - red[0]);
  __syncthreads();
  if (tid == 0) {
    float sum = 0.f;
    for (int s = 0; s < S_; ++s) sum += sc[s];
    red[1] = 1.f / sum;
  }
  __syncthreads();
  const float* vp = Vh + (size_t)b * S_ * H_ + nh * HD_ + tid;
  float acc = 0.f;
  for (int s = 0; s < S_; ++s) acc += sc[s] * vp[(size_t)s * H_];
  ctxb[(size_t)b * H_ + nh * HD_ + tid] = (__bf16)(acc * red[1]);
}

extern "C" void kernel_launch(void* const* d_in, const int* in_sizes, int n_in,
                              void* d_out, int out_size, void* d_ws, size_t ws_size,
                              hipStream_t stream) {
  (void)in_sizes; (void)n_in; (void)out_size; (void)ws_size;

  const float* encoder_out = (const float*)d_in[0];
  const int*   captions    = (const int*)  d_in[1];
  const float* embed_table = (const float*)d_in[2];
  const float* w_ih        = (const float*)d_in[3];
  const float* w_hh        = (const float*)d_in[4];
  const float* b_ih        = (const float*)d_in[5];
  const float* b_hh        = (const float*)d_in[6];
  const float* in_proj_w   = (const float*)d_in[7];
  const float* in_proj_b   = (const float*)d_in[8];
  const float* out_proj_w  = (const float*)d_in[9];
  const float* out_proj_b  = (const float*)d_in[10];
  const float* fc_w        = (const float*)d_in[11];
  const float* fc_b        = (const float*)d_in[12];
  float* out = (float*)d_out;

  // ---- workspace carve-out (256B aligned) ----
  char* ws = (char*)d_ws;
  size_t off = 0;
  auto carve = [&](size_t bytes) -> void* {
    off = (off + 255) & ~(size_t)255;
    void* p = ws + off;
    off += bytes;
    return p;
  };
  __bf16* fcw_bf  = (__bf16*)carve((size_t)V_ * H_ * 2);
  __bf16* wih_bf  = (__bf16*)carve((size_t)L_ * 3 * H_ * E_ * 2);
  __bf16* whh_bf  = (__bf16*)carve((size_t)L_ * 3 * H_ * H_ * 2);
  __bf16* inp_bf  = (__bf16*)carve((size_t)3 * H_ * H_ * 2);
  __bf16* outp_bf = (__bf16*)carve((size_t)H_ * H_ * 2);
  __bf16* enc_bf  = (__bf16*)carve((size_t)B_ * S_ * H_ * 2);
  float*  Kh      = (float*) carve((size_t)B_ * S_ * H_ * 4);
  float*  Vh      = (float*) carve((size_t)B_ * S_ * H_ * 4);
  float*  hf      = (float*) carve((size_t)L_ * B_ * H_ * 4);
  __bf16* hb      = (__bf16*)carve((size_t)L_ * B_ * H_ * 2);
  __bf16* xb      = (__bf16*)carve((size_t)B_ * E_ * 2);
  float*  gi      = (float*) carve((size_t)B_ * 3 * H_ * 4);
  float*  gh      = (float*) carve((size_t)B_ * 3 * H_ * 4);
  float*  qf      = (float*) carve((size_t)B_ * H_ * 4);
  __bf16* ctxb    = (__bf16*)carve((size_t)B_ * H_ * 2);
  float*  aof     = (float*) carve((size_t)B_ * H_ * 4);
  __bf16* aob     = (__bf16*)carve((size_t)B_ * H_ * 2);

  auto cvt = [&](const float* s, __bf16* d, size_t n) {
    cvt_f32_bf16<<<(unsigned)((n + 255) / 256), 256, 0, stream>>>(s, d, (int)n);
  };
  auto gemm = [&](const __bf16* A, int lda, const __bf16* Bw, int ldb,
                  const float* bias, float* C, int ldc, __bf16* Cb, int ldcb,
                  int M, int N, int K) {
    dim3 grid((unsigned)(N / 64), (unsigned)(M / 16), 1);
    gemm_bf16_wmma<<<grid, 32, 0, stream>>>(A, lda, Bw, ldb, bias, C, ldc, Cb, ldcb, K);
  };
  auto gemm_async = [&](const __bf16* A, int lda, const __bf16* Bw, int ldb,
                        const float* bias, float* C, int ldc, int M, int N, int K) {
    dim3 grid((unsigned)(N / 64), (unsigned)(M / 64), 1);
    gemm_bf16_wmma_async<<<grid, 128, 0, stream>>>(A, lda, Bw, ldb, bias, C, ldc, K);
  };

  // ---- one-time (per launch) precompute ----
  cvt(fc_w,        fcw_bf,  (size_t)V_ * H_);
  cvt(w_ih,        wih_bf,  (size_t)L_ * 3 * H_ * E_);
  cvt(w_hh,        whh_bf,  (size_t)L_ * 3 * H_ * H_);
  cvt(in_proj_w,   inp_bf,  (size_t)3 * H_ * H_);
  cvt(out_proj_w,  outp_bf, (size_t)H_ * H_);
  cvt(encoder_out, enc_bf,  (size_t)B_ * S_ * H_);
  zero_state<<<(L_ * B_ * H_) / 256, 256, 0, stream>>>(hf, hb, L_ * B_ * H_);

  // K/V encoder projections (timestep-invariant): (B*S, H) = enc @ Wk^T + bk
  gemm_async(enc_bf, H_, inp_bf + (size_t)1 * H_ * H_, H_, in_proj_b + H_,
             Kh, H_, B_ * S_, H_, H_);
  gemm_async(enc_bf, H_, inp_bf + (size_t)2 * H_ * H_, H_, in_proj_b + 2 * H_,
             Vh, H_, B_ * S_, H_, H_);

  // ---- time loop ----
  for (int t = 0; t < T_; ++t) {
    embed_gather<<<(B_ * E_) / 256, 256, 0, stream>>>(embed_table, captions, xb, t);

    for (int l = 0; l < L_; ++l) {
      const __bf16* xin = (l == 0) ? xb : (hb + (size_t)(l - 1) * B_ * H_);
      const __bf16* wi  = wih_bf + (size_t)l * 3 * H_ * E_;
      const __bf16* wh  = whh_bf + (size_t)l * 3 * H_ * H_;
      gemm(xin,                      E_, wi, E_, b_ih + (size_t)l * 3 * H_,
           gi, 3 * H_, nullptr, 0, B_, 3 * H_, E_);
      gemm(hb + (size_t)l * B_ * H_, H_, wh, H_, b_hh + (size_t)l * 3 * H_,
           gh, 3 * H_, nullptr, 0, B_, 3 * H_, H_);
      gru_gate<<<(B_ * H_) / 256, 256, 0, stream>>>(
          gi, gh, hf + (size_t)l * B_ * H_, hb + (size_t)l * B_ * H_);
    }

    // q = h3 @ Wq^T + bq
    gemm(hb + (size_t)2 * B_ * H_, H_, inp_bf, H_, in_proj_b,
         qf, H_, nullptr, 0, B_, H_, H_);

    // attention: softmax(q.K^T / sqrt(HD)) . V  -> ctx (bf16)
    attn_softmax_ctx<<<B_ * NH_, S_, 0, stream>>>(qf, Kh, Vh, ctxb);

    // ao = ctx @ out_proj_w^T + b  (fp32 + bf16 mirror)
    gemm(ctxb, H_, outp_bf, H_, out_proj_b, aof, H_, aob, H_, B_, H_, H_);

    // logits[b, t, :] = ao @ fc_w^T + fc_b, async-staged B (fc_w dominates)
    gemm_async(aob, H_, fcw_bf, H_, fc_b, out + (size_t)t * V_, T_ * V_,
               B_, V_, H_);
  }
}
